// StackGRU_24515673325785
// MI455X (gfx1250) — compile-verified, matrix operand order checked by reading, not run on hardware
//
#include <hip/hip_runtime.h>
#include <hip/hip_bf16.h>

// ---------------------------------------------------------------------------
// Stacked 2-layer GRU scan, persistent-kernel design for MI455X (gfx1250).
//   B=64, T=1024, I=64, H=512, O=24
// - 16 persistent workgroups x 256 threads (8 wave32 waves each), one per WGP.
// - Each WG owns a 32-unit hidden slice; its r/z/n weight rows are staged into
//   LDS once (300 KB of the 320 KB WGP LDS), then every timestep's gate GEMMs
//   run v_wmma_f32_16x16x32_bf16 with B-fragments from ds_load_b128.
// - h state is double-buffered bf16 in global; grid barrier 2x per step.
// ---------------------------------------------------------------------------

#define B_ 64
#define T_ 1024
#define I_ 64
#define H_ 512
#define O_ 24
#define NWG 16
#define HSLICE (H_ / NWG)   // 32 hidden units per workgroup
#define GROWS (3 * HSLICE)  // 96 staged weight rows per matrix (r,z,n)

typedef unsigned short bf_t; // bf16 stored as raw u16 (portable, no __bf16 math)
typedef __attribute__((ext_vector_type(16))) __bf16        v16bf;
typedef __attribute__((ext_vector_type(8)))  float         v8f;
typedef __attribute__((ext_vector_type(4)))  unsigned int  v4u;
typedef __attribute__((ext_vector_type(8)))  unsigned int  v8u;

// LDS layout (elements): wih1[96*64] | whh1[96*512] | wih2[96*512] | whh2[96*512]
#define LDS_WIH1 0
#define LDS_WHH1 (GROWS * I_)
#define LDS_WIH2 (LDS_WHH1 + GROWS * H_)
#define LDS_WHH2 (LDS_WIH2 + GROWS * H_)
#define LDS_ELEMS (LDS_WHH2 + GROWS * H_)          // 153,600 elems = 300 KB
#define LDS_BYTES (LDS_ELEMS * 2)

// ---------------- bf16 <-> f32 (round-to-nearest-even) ---------------------
__device__ __forceinline__ bf_t f2bf(float f) {
  unsigned u = __builtin_bit_cast(unsigned, f);
  u += 0x7FFFu + ((u >> 16) & 1u);
  return (bf_t)(u >> 16);
}
__device__ __forceinline__ float bf2f(bf_t b) {
  return __builtin_bit_cast(float, ((unsigned)b) << 16);
}

// -------- load one wave's 16x32 bf16 A/B fragment from row-major data ------
// CDNA5 16-bit operand layout: lane L holds row (L&15); its 16 elements are
// K = [8*half .. +7] and [16+8*half .. +7]  (half = L>>4): two b128 chunks.
__device__ __forceinline__ v16bf load_frag_row(const bf_t* row, int k0, int half) {
  const bf_t* p = row + k0 + 8 * half;
  v4u c0 = *(const v4u*)(p);
  v4u c1 = *(const v4u*)(p + 16);
  v8u r;
  r[0] = c0[0]; r[1] = c0[1]; r[2] = c0[2]; r[3] = c0[3];
  r[4] = c1[0]; r[5] = c1[1]; r[6] = c1[2]; r[7] = c1[3];
  return __builtin_bit_cast(v16bf, r);
}

__device__ __forceinline__ v8f wmma_bf16(v16bf a, v16bf b, v8f c) {
  return __builtin_amdgcn_wmma_f32_16x16x32_bf16(
      /*neg_a=*/false, a, /*neg_b=*/false, b,
      /*c_mod=*/(short)0, c, /*reuse_a=*/false, /*reuse_b=*/false);
}

// Accumulate 3 gate tiles: A[16xK] (global) x staged W rows (LDS).
// wrow points at this lane's r-gate row inside the staged slice; gate stride
// is HSLICE rows. `zero` is an opaque per-iteration 0 that defeats LICM.
__device__ __forceinline__ void gemm3_lds(v8f& g0, v8f& g1, v8f& g2,
                                          const bf_t* Atile, size_t astride,
                                          const bf_t* wrow, int K,
                                          int lane, unsigned zero) {
  const int half = lane >> 4;
  const bf_t* arow = Atile + (size_t)(lane & 15) * astride;
  const bf_t* w0 = wrow + zero;
  for (int k0 = 0; k0 < K; k0 += 32) {
    v16bf a = load_frag_row(arow, k0, half);
    g0 = wmma_bf16(a, load_frag_row(w0, k0, half), g0);
    g1 = wmma_bf16(a, load_frag_row(w0 + HSLICE * K, k0, half), g1);
    g2 = wmma_bf16(a, load_frag_row(w0 + 2 * HSLICE * K, k0, half), g2);
  }
}

__device__ __forceinline__ float sigmoidf_(float x) {
  return 1.0f / (1.0f + __expf(-x));
}

// GRU elementwise update on the wave's accumulator tile; writes h slice.
__device__ __forceinline__ void gru_update(const v8f& ar, const v8f& az,
                                           const v8f& ain, const v8f& ahn,
                                           const float* bih, const float* bhh,
                                           const bf_t* hread, bf_t* hwrite,
                                           int m_base, int jc, int half) {
  const float br  = bih[jc] + bhh[jc];
  const float bz  = bih[H_ + jc] + bhh[H_ + jc];
  const float bin = bih[2 * H_ + jc];
  const float bhn = bhh[2 * H_ + jc];
#pragma unroll
  for (int i = 0; i < 8; ++i) {
    const int m = m_base + 8 * half + i;   // D layout: M = vgpr + 8*(lane>>4)
    const float r = sigmoidf_(ar[i] + br);
    const float z = sigmoidf_(az[i] + bz);
    const float n = tanhf(ain[i] + bin + r * (ahn[i] + bhn));
    const float hp = bf2f(hread[(size_t)m * H_ + jc]);
    hwrite[(size_t)m * H_ + jc] = f2bf((1.0f - z) * n + z * hp);
  }
}

// ---------------- grid-wide barrier (monotonic epoch counter) --------------
__device__ __forceinline__ void grid_barrier(unsigned* cnt, unsigned target) {
  __builtin_amdgcn_fence(__ATOMIC_RELEASE, "agent");
  __syncthreads();
  if (threadIdx.x == 0) {
    __hip_atomic_fetch_add(cnt, 1u, __ATOMIC_RELAXED, __HIP_MEMORY_SCOPE_AGENT);
    while (__hip_atomic_load(cnt, __ATOMIC_RELAXED, __HIP_MEMORY_SCOPE_AGENT) <
           target) {
      __builtin_amdgcn_s_sleep(8);
    }
  }
  __syncthreads();
  __builtin_amdgcn_fence(__ATOMIC_ACQUIRE, "agent");
}

// Stage this WG's 96 weight rows (r,z,n x 32 local units) into LDS.
__device__ __forceinline__ void stage_slice(bf_t* dst, const bf_t* src, int K,
                                            int wg) {
  const int cpr = K / 8;                      // 16-byte chunks per row
  for (int c = threadIdx.x; c < GROWS * cpr; c += 256) {
    const int dr = c / cpr;
    const int kk = (c - dr * cpr) * 8;
    const int gate = dr >> 5, lr = dr & 31;
    const bf_t* s = src + (size_t)(gate * H_ + wg * HSLICE + lr) * K + kk;
    *(v4u*)(dst + dr * K + kk) = *(const v4u*)s;
  }
}

// --------------------------- prep kernels ----------------------------------
__global__ void cvt_f32_bf16(const float* __restrict__ src,
                             bf_t* __restrict__ dst, int n) {
  int i = blockIdx.x * blockDim.x + threadIdx.x;
  const int stride = gridDim.x * blockDim.x;
  for (; i < n; i += stride) dst[i] = f2bf(src[i]);
}

__global__ void init_state(bf_t* h1, bf_t* h2, unsigned* cnt, int n) {
  int i = blockIdx.x * blockDim.x + threadIdx.x;
  const int stride = gridDim.x * blockDim.x;
  for (; i < n; i += stride) { h1[i] = 0; h2[i] = 0; }
  if (blockIdx.x == 0 && threadIdx.x == 0) *cnt = 0u;
}

// ------------------------- persistent GRU kernel ---------------------------
__global__ __launch_bounds__(256, 1) void gru_persistent(
    const bf_t* __restrict__ xbf,
    const bf_t* __restrict__ wih1, const bf_t* __restrict__ whh1,
    const bf_t* __restrict__ wih2, const bf_t* __restrict__ whh2,
    const bf_t* __restrict__ wo1, const bf_t* __restrict__ wo2,
    const float* __restrict__ bih1, const float* __restrict__ bhh1,
    const float* __restrict__ bih2, const float* __restrict__ bhh2,
    const float* __restrict__ bo1, const float* __restrict__ bo2,
    bf_t* __restrict__ h1buf, bf_t* __restrict__ h2buf,
    float* __restrict__ out, unsigned* __restrict__ cnt) {
  extern __shared__ bf_t smem[];
  bf_t* wih1s = smem + LDS_WIH1;
  bf_t* whh1s = smem + LDS_WHH1;
  bf_t* wih2s = smem + LDS_WIH2;
  bf_t* whh2s = smem + LDS_WHH2;

  const int wg = blockIdx.x;
  const int lane = threadIdx.x & 31;
  const int wave = threadIdx.x >> 5;
  const int m_tile = (wave & 3) * 16;        // batch-row tile (B=64 -> 4 tiles)
  const int n_half = wave >> 2;              // which 16-col half of the slice
  const int jc = wg * HSLICE + n_half * 16 + (lane & 15);
  const int half = lane >> 4;
  // This lane's r-gate row inside each staged slice:
  const int lrow = n_half * 16 + (lane & 15);
  unsigned bar = 0;

  // ---- one-time weight staging into LDS ----
  stage_slice(wih1s, wih1, I_, wg);
  stage_slice(whh1s, whh1, H_, wg);
  stage_slice(wih2s, wih2, H_, wg);
  stage_slice(whh2s, whh2, H_, wg);
  __syncthreads();

  for (int t = 0; t < T_; ++t) {
    // Opaque zero, re-materialized every iteration: keeps the LDS/global
    // weight-fragment loads inside the loop (no LICM -> no register spills).
    unsigned zero;
    asm volatile("s_mov_b32 %0, 0" : "=s"(zero));

    const bf_t* h1r = h1buf + (size_t)(t & 1) * B_ * H_;
    bf_t*       h1w = h1buf + (size_t)((t + 1) & 1) * B_ * H_;
    const bf_t* h2r = h2buf + (size_t)(t & 1) * B_ * H_;
    bf_t*       h2w = h2buf + (size_t)((t + 1) & 1) * B_ * H_;

    // ---- layer 1: h1 = GRU(x_t, h1) ----
    {
      v8f ar = {}, az = {}, ain = {}, ahn = {};
      gemm3_lds(ar, az, ain,
                xbf + (size_t)m_tile * T_ * I_ + (size_t)t * I_, (size_t)T_ * I_,
                wih1s + lrow * I_, I_, lane, zero);
      gemm3_lds(ar, az, ahn, h1r + (size_t)m_tile * H_, (size_t)H_,
                whh1s + lrow * H_, H_, lane, zero);
      gru_update(ar, az, ain, ahn, bih1, bhh1, h1r, h1w, m_tile, jc, half);
    }
    bar++; grid_barrier(cnt, bar * NWG);

    // ---- layer 2: h2 = GRU(h1, h2) ----
    {
      v8f ar = {}, az = {}, ain = {}, ahn = {};
      gemm3_lds(ar, az, ain, h1w + (size_t)m_tile * H_, (size_t)H_,
                wih2s + lrow * H_, H_, lane, zero);
      gemm3_lds(ar, az, ahn, h2r + (size_t)m_tile * H_, (size_t)H_,
                whh2s + lrow * H_, H_, lane, zero);
      gru_update(ar, az, ain, ahn, bih2, bhh2, h2r, h2w, m_tile, jc, half);
    }
    bar++; grid_barrier(cnt, bar * NWG);

    // ---- output head (WG0 only): out_t = tanh(h1 Wo1^T+b) + tanh(h2 Wo2^T+b)
    if (wg == 0) {
      const int oc = n_half * 16 + (lane & 15);          // 0..31 (>=24 unused)
      const int orow = oc < O_ ? oc : O_ - 1;            // clamp, never stored
      v8f a1 = {}, a2 = {};
      const bf_t* A1 = h1w + (size_t)(m_tile + (lane & 15)) * H_;
      const bf_t* A2 = h2w + (size_t)(m_tile + (lane & 15)) * H_;
      const bf_t* W1 = wo1 + (size_t)orow * H_ + zero;
      const bf_t* W2 = wo2 + (size_t)orow * H_ + zero;
      for (int k0 = 0; k0 < H_; k0 += 32) {
        a1 = wmma_bf16(load_frag_row(A1, k0, half),
                       load_frag_row(W1, k0, half), a1);
        a2 = wmma_bf16(load_frag_row(A2, k0, half),
                       load_frag_row(W2, k0, half), a2);
      }
      if (oc < O_) {
        const float c1 = bo1[oc], c2 = bo2[oc];
#pragma unroll
        for (int i = 0; i < 8; ++i) {
          const int m = m_tile + 8 * half + i;
          out[(size_t)m * T_ * O_ + (size_t)t * O_ + oc] =
              tanhf(a1[i] + c1) + tanhf(a2[i] + c2);
        }
      }
    }
  }
}

// ------------------------------- launcher ----------------------------------
extern "C" void kernel_launch(void* const* d_in, const int* in_sizes, int n_in,
                              void* d_out, int out_size, void* d_ws,
                              size_t ws_size, hipStream_t stream) {
  const float* x    = (const float*)d_in[0];
  const float* wih1 = (const float*)d_in[1];
  const float* whh1 = (const float*)d_in[2];
  const float* bih1 = (const float*)d_in[3];
  const float* bhh1 = (const float*)d_in[4];
  const float* wih2 = (const float*)d_in[5];
  const float* whh2 = (const float*)d_in[6];
  const float* bih2 = (const float*)d_in[7];
  const float* bhh2 = (const float*)d_in[8];
  const float* wo1  = (const float*)d_in[9];
  const float* bo1  = (const float*)d_in[10];
  const float* wo2  = (const float*)d_in[11];
  const float* bo2  = (const float*)d_in[12];
  float* out = (float*)d_out;

  // d_ws layout (bytes). Total ~13.7 MB.
  char* ws = (char*)d_ws;
  const size_t nX    = (size_t)B_ * T_ * I_;     // 4,194,304
  const size_t nWih1 = (size_t)3 * H_ * I_;      //    98,304
  const size_t nWhh  = (size_t)3 * H_ * H_;      //   786,432
  const size_t nWo   = (size_t)O_ * H_;          //    12,288
  const size_t nH    = (size_t)2 * B_ * H_;      //    65,536 (double buffer)

  size_t off = 0;
  bf_t* xbf   = (bf_t*)(ws + off); off += nX * 2;
  bf_t* wih1b = (bf_t*)(ws + off); off += nWih1 * 2;
  bf_t* whh1b = (bf_t*)(ws + off); off += nWhh * 2;
  bf_t* wih2b = (bf_t*)(ws + off); off += nWhh * 2;
  bf_t* whh2b = (bf_t*)(ws + off); off += nWhh * 2;
  bf_t* wo1b  = (bf_t*)(ws + off); off += nWo * 2;
  bf_t* wo2b  = (bf_t*)(ws + off); off += nWo * 2;
  bf_t* h1b   = (bf_t*)(ws + off); off += nH * 2;
  bf_t* h2b   = (bf_t*)(ws + off); off += nH * 2;
  unsigned* cnt = (unsigned*)(ws + off); off += 256;
  (void)ws_size; (void)in_sizes; (void)n_in; (void)out_size;

  cvt_f32_bf16<<<1024, 256, 0, stream>>>(x,    xbf,   (int)nX);
  cvt_f32_bf16<<<128,  256, 0, stream>>>(wih1, wih1b, (int)nWih1);
  cvt_f32_bf16<<<512,  256, 0, stream>>>(whh1, whh1b, (int)nWhh);
  cvt_f32_bf16<<<512,  256, 0, stream>>>(whh2, whh2b, (int)nWhh);
  cvt_f32_bf16<<<512,  256, 0, stream>>>(wih2, wih2b, (int)nWhh);
  cvt_f32_bf16<<<32,   256, 0, stream>>>(wo1,  wo1b,  (int)nWo);
  cvt_f32_bf16<<<32,   256, 0, stream>>>(wo2,  wo2b,  (int)nWo);
  init_state<<<64, 256, 0, stream>>>(h1b, h2b, cnt, (int)nH);

  // Allow 300 KB of dynamic LDS (one workgroup per WGP).
  (void)hipFuncSetAttribute((const void*)gru_persistent,
                            hipFuncAttributeMaxDynamicSharedMemorySize,
                            LDS_BYTES);

  gru_persistent<<<NWG, 256, LDS_BYTES, stream>>>(
      xbf, wih1b, whh1b, wih2b, whh2b, wo1b, wo2b,
      bih1, bhh1, bih2, bhh2, bo1, bo2, h1b, h2b, out, cnt);
}
